// ScaledBinaryConv1D_91113436217715
// MI455X (gfx1250) — compile-verified, matrix operand order checked by reading, not run on hardware
//
#include <hip/hip_runtime.h>

// ---------------------------------------------------------------------------
// y[M,N] = (x[M,K] @ sign[K,N]) * softplus(scale)
// M = B*S = 4096, K = 4096, N = 4096, all f32 in memory.
//
// Compute-bound GEMM (715 FLOP/byte >> machine balance ~200 FLOP/byte at
// f16-WMMA peak). Strategy:
//   * f16 hi/lo split of x (Markidis two-term), sign is exactly +/-1 -> exact
//     in f16; V_WMMA_F32_16X16X32_F16 with f32 accumulation (~22 mantissa
//     bits recovered, near-f32 results).
//   * Pre-pass (bandwidth-bound, ~10us at 23.3 TB/s) converts x -> xhi/xlo
//     f16 and sign -> signT f16 (pre-transposed) into d_ws. This removes the
//     32x-redundant per-tile conversion VALU and the transpose scatter from
//     the GEMM inner loop, and makes staging pure 16B copies -> eligible for
//     CDNA5 GLOBAL_LOAD_ASYNC_TO_LDS_B128 (ASYNCcnt) DMA.
//   * GEMM: 128x128 block tile, 8 waves (wave32) in 4x2, 32x64 per wave,
//     BK=32, double-buffered LDS, one barrier per k-step; async staging of
//     tile k+1 overlaps the WMMAs of tile k.
//   * Fallback (ws too small): fused-conversion
//     kernel with register-staged software pipeline.
// ---------------------------------------------------------------------------

typedef __attribute__((ext_vector_type(16))) _Float16 v16h;
typedef __attribute__((ext_vector_type(8)))  _Float16 v8h;
typedef __attribute__((ext_vector_type(4)))  _Float16 v4h;
typedef __attribute__((ext_vector_type(8)))  float    v8f;
typedef __attribute__((ext_vector_type(4)))  int      v4i;

#define M_TOTAL 4096
#define N_TOTAL 4096
#define K_TOTAL 4096

#define BM 128
#define BN 128
#define BK 32
#define NKSTEPS (K_TOTAL / BK)
#define LDSS 48   // f16 per LDS row: 32 data + 16 pad (96B rows, 16B-aligned)

#if defined(__has_builtin)
#if __has_builtin(__builtin_amdgcn_global_load_async_to_lds_b128) && \
    __has_builtin(__builtin_amdgcn_s_wait_asynccnt)
#define USE_ASYNC_COPY 1
#endif
#endif
#ifndef USE_ASYNC_COPY
#define USE_ASYNC_COPY 0
#endif

// 16-byte global->LDS copy: async DMA if available, else load+store.
__device__ __forceinline__ void cp16(const _Float16* g, _Float16* l) {
#if USE_ASYNC_COPY
    // Param pointee type is int4 (per clang diagnostic); qualify address
    // spaces explicitly: AS1 (global) source, AS3 (LDS) destination.
    __builtin_amdgcn_global_load_async_to_lds_b128(
        (__attribute__((address_space(1))) v4i*)(g),
        (__attribute__((address_space(3))) v4i*)(l),
        0, 0);
#else
    *reinterpret_cast<v8h*>(l) = *reinterpret_cast<const v8h*>(g);
#endif
}

__device__ __forceinline__ void cp_drain() {
#if USE_ASYNC_COPY
    __builtin_amdgcn_s_wait_asynccnt(0);
#endif
}

__device__ __forceinline__ v16h frag_from_lds(const _Float16* row, int half) {
    // Per ISA 16-bit A/B layout: lane holds k = half*8 + [0,8) in VGPRs 0-3
    // and k = 16 + half*8 + [0,8) in VGPRs 4-7 -> two contiguous ds_load_b128.
    const _Float16* p = row + half * 8;
    v8h lo = *reinterpret_cast<const v8h*>(p);
    v8h hi = *reinterpret_cast<const v8h*>(p + 16);
    return __builtin_shufflevector(lo, hi, 0, 1, 2, 3, 4, 5, 6, 7,
                                           8, 9, 10, 11, 12, 13, 14, 15);
}

__device__ __forceinline__ float softplusf(float sc) {
    return (sc > 20.f) ? sc : log1pf(expf(sc));
}

// ===========================================================================
// Pre-pass 1: x (f32) -> xhi, xlo (f16), elementwise. 4 f32 per thread.
// ===========================================================================
__global__ __launch_bounds__(256)
void ConvertX(const float* __restrict__ X,
              _Float16* __restrict__ Xhi, _Float16* __restrict__ Xlo) {
    size_t idx = ((size_t)blockIdx.x * 256 + threadIdx.x) * 4;
    float4 v = *reinterpret_cast<const float4*>(X + idx);
    _Float16 h0 = (_Float16)v.x, h1 = (_Float16)v.y;
    _Float16 h2 = (_Float16)v.z, h3 = (_Float16)v.w;
    v4h H = {h0, h1, h2, h3};
    v4h L = {(_Float16)(v.x - (float)h0), (_Float16)(v.y - (float)h1),
             (_Float16)(v.z - (float)h2), (_Float16)(v.w - (float)h3)};
    *reinterpret_cast<v4h*>(Xhi + idx) = H;
    *reinterpret_cast<v4h*>(Xlo + idx) = L;
}

// ===========================================================================
// Pre-pass 2: sign (K x N f32) -> signT (N x K f16), 64x64 LDS tile transpose.
// ===========================================================================
__global__ __launch_bounds__(256)
void TransposeSign(const float* __restrict__ Sgn, _Float16* __restrict__ ST) {
    __shared__ _Float16 t[64][80];           // 80: pad, keeps 16B row alignment
    const int tid = threadIdx.x;
    const int k0 = blockIdx.y * 64;          // source row block (K)
    const int n0 = blockIdx.x * 64;          // source col block (N)

    // Load 64x64 f32 (coalesced along n), store transposed f16 into LDS.
#pragma unroll
    for (int i = 0; i < 4; ++i) {
        int flat = i * 256 + tid;            // 0..1023 float4 chunks
        int r  = flat >> 4;                  // k-row, 16 float4 per row
        int c4 = (flat & 15) << 2;           // n-col
        float4 v = *reinterpret_cast<const float4*>(
            Sgn + (size_t)(k0 + r) * N_TOTAL + n0 + c4);
        t[c4 + 0][r] = (_Float16)v.x;
        t[c4 + 1][r] = (_Float16)v.y;
        t[c4 + 2][r] = (_Float16)v.z;
        t[c4 + 3][r] = (_Float16)v.w;
    }
    __syncthreads();

    // Write 64 rows of 64 f16 (128B each) to signT, coalesced b128 stores.
#pragma unroll
    for (int j = 0; j < 2; ++j) {
        int flat = j * 256 + tid;            // 0..511 v8h chunks
        int orow = flat >> 3;                // n local
        int oc   = (flat & 7) << 3;          // k local
        *reinterpret_cast<v8h*>(ST + (size_t)(n0 + orow) * K_TOTAL + k0 + oc) =
            *reinterpret_cast<const v8h*>(&t[orow][oc]);
    }
}

// ===========================================================================
// Main GEMM over pre-converted f16 operands (async-staged LDS pipeline).
// ===========================================================================
__global__ __launch_bounds__(256)
void GemmF16Wmma(const _Float16* __restrict__ Xhi,
                 const _Float16* __restrict__ Xlo,
                 const _Float16* __restrict__ ST,
                 const float* __restrict__ ScaleP,
                 float* __restrict__ Out) {
    __shared__ _Float16 sAhi[2][BM][LDSS];
    __shared__ _Float16 sAlo[2][BM][LDSS];
    __shared__ _Float16 sBt [2][BN][LDSS];

    const int tid    = threadIdx.x;
    const int lane   = tid & 31;
    const int wid    = tid >> 5;
    const int wave_m = wid & 3;
    const int wave_n = wid >> 2;
    const int half   = lane >> 4;
    const int lane16 = lane & 15;

    const int m0 = blockIdx.y * BM;
    const int n0 = blockIdx.x * BN;

    v8f acc[2][4];
#pragma unroll
    for (int tm = 0; tm < 2; ++tm)
#pragma unroll
        for (int tn = 0; tn < 4; ++tn)
            acc[tm][tn] = (v8f){0.f, 0.f, 0.f, 0.f, 0.f, 0.f, 0.f, 0.f};

    // Stage one 128x32 f16 tile per operand: 512 16B chunks, 2 per thread.
    auto stage = [&](int buf, int kb) {
#pragma unroll
        for (int j = 0; j < 2; ++j) {
            int c   = j * 256 + tid;        // 0..511
            int row = c >> 2;               // 4 chunks of 8 f16 per 32-wide row
            int col = (c & 3) << 3;
            cp16(Xhi + (size_t)(m0 + row) * K_TOTAL + kb + col,
                 &sAhi[buf][row][col]);
            cp16(Xlo + (size_t)(m0 + row) * K_TOTAL + kb + col,
                 &sAlo[buf][row][col]);
            cp16(ST  + (size_t)(n0 + row) * K_TOTAL + kb + col,
                 &sBt[buf][row][col]);
        }
    };

    stage(0, 0);
    cp_drain();
    __syncthreads();

    for (int it = 0; it < NKSTEPS; ++it) {
        const int buf = it & 1;
        if (it + 1 < NKSTEPS) stage(buf ^ 1, (it + 1) * BK);  // DMA overlaps WMMA

        v16h aHi[2], aLo[2], bF[4];
#pragma unroll
        for (int tm = 0; tm < 2; ++tm) {
            int r = wave_m * 32 + tm * 16 + lane16;
            aHi[tm] = frag_from_lds(&sAhi[buf][r][0], half);
            aLo[tm] = frag_from_lds(&sAlo[buf][r][0], half);
        }
#pragma unroll
        for (int tn = 0; tn < 4; ++tn) {
            int r = wave_n * 64 + tn * 16 + lane16;
            bF[tn] = frag_from_lds(&sBt[buf][r][0], half);
        }

#pragma unroll
        for (int tm = 0; tm < 2; ++tm)
#pragma unroll
            for (int tn = 0; tn < 4; ++tn) {
                acc[tm][tn] = __builtin_amdgcn_wmma_f32_16x16x32_f16(
                    false, aHi[tm], false, bF[tn], (short)0, acc[tm][tn],
                    false, false);
                acc[tm][tn] = __builtin_amdgcn_wmma_f32_16x16x32_f16(
                    false, aLo[tm], false, bF[tn], (short)0, acc[tm][tn],
                    false, false);
            }

        if (it + 1 < NKSTEPS) cp_drain();  // my async stores done before barrier
        __syncthreads();
    }

    float sp = softplusf(ScaleP[0]);
#pragma unroll
    for (int tm = 0; tm < 2; ++tm)
#pragma unroll
        for (int tn = 0; tn < 4; ++tn) {
            int col = n0 + wave_n * 64 + tn * 16 + lane16;
#pragma unroll
            for (int r = 0; r < 8; ++r) {
                int row = m0 + wave_m * 32 + tm * 16 + half * 8 + r;
                Out[(size_t)row * N_TOTAL + col] = acc[tm][tn][r] * sp;
            }
        }
}

// ===========================================================================
// Fallback: fused-conversion GEMM (no workspace needed).
// ===========================================================================
__global__ __launch_bounds__(256)
void GemmFusedWmma(const float* __restrict__ X,
                   const float* __restrict__ Sgn,
                   const float* __restrict__ ScaleP,
                   float* __restrict__ Out) {
    __shared__ _Float16 sAhi[2][BM][LDSS];
    __shared__ _Float16 sAlo[2][BM][LDSS];
    __shared__ _Float16 sBt [2][BN][LDSS];

    const int tid    = threadIdx.x;
    const int lane   = tid & 31;
    const int wid    = tid >> 5;
    const int wave_m = wid & 3;
    const int wave_n = wid >> 2;
    const int half   = lane >> 4;
    const int lane16 = lane & 15;

    const int m0 = blockIdx.y * BM;
    const int n0 = blockIdx.x * BN;

    v8f acc[2][4];
#pragma unroll
    for (int tm = 0; tm < 2; ++tm)
#pragma unroll
        for (int tn = 0; tn < 4; ++tn)
            acc[tm][tn] = (v8f){0.f, 0.f, 0.f, 0.f, 0.f, 0.f, 0.f, 0.f};

    float4 aR[4], bR[4];

    auto globalLoad = [&](int kb) {
#pragma unroll
        for (int i = 0; i < 4; ++i) {
            int flat = i * 256 + tid;
            {
                int row = flat >> 3, c4 = (flat & 7) << 2;
                aR[i] = *reinterpret_cast<const float4*>(
                    X + (size_t)(m0 + row) * K_TOTAL + kb + c4);
            }
            {
                int kr = flat >> 5, c4 = (flat & 31) << 2;
                bR[i] = *reinterpret_cast<const float4*>(
                    Sgn + (size_t)(kb + kr) * N_TOTAL + n0 + c4);
            }
        }
    };

    auto ldsStore = [&](int buf) {
#pragma unroll
        for (int i = 0; i < 4; ++i) {
            int flat = i * 256 + tid;
            {
                int row = flat >> 3, c4 = (flat & 7) << 2;
                float4 v = aR[i];
                _Float16 h0 = (_Float16)v.x, h1 = (_Float16)v.y;
                _Float16 h2 = (_Float16)v.z, h3 = (_Float16)v.w;
                sAhi[buf][row][c4 + 0] = h0;
                sAhi[buf][row][c4 + 1] = h1;
                sAhi[buf][row][c4 + 2] = h2;
                sAhi[buf][row][c4 + 3] = h3;
                sAlo[buf][row][c4 + 0] = (_Float16)(v.x - (float)h0);
                sAlo[buf][row][c4 + 1] = (_Float16)(v.y - (float)h1);
                sAlo[buf][row][c4 + 2] = (_Float16)(v.z - (float)h2);
                sAlo[buf][row][c4 + 3] = (_Float16)(v.w - (float)h3);
            }
            {
                int kr = flat >> 5, c4 = (flat & 31) << 2;
                float4 v = bR[i];
                sBt[buf][c4 + 0][kr] = (_Float16)v.x;
                sBt[buf][c4 + 1][kr] = (_Float16)v.y;
                sBt[buf][c4 + 2][kr] = (_Float16)v.z;
                sBt[buf][c4 + 3][kr] = (_Float16)v.w;
            }
        }
    };

    globalLoad(0);
    ldsStore(0);
    __syncthreads();

    for (int it = 0; it < NKSTEPS; ++it) {
        const int buf = it & 1;
        if (it + 1 < NKSTEPS) globalLoad((it + 1) * BK);

        v16h aHi[2], aLo[2], bF[4];
#pragma unroll
        for (int tm = 0; tm < 2; ++tm) {
            int r = wave_m * 32 + tm * 16 + lane16;
            aHi[tm] = frag_from_lds(&sAhi[buf][r][0], half);
            aLo[tm] = frag_from_lds(&sAlo[buf][r][0], half);
        }
#pragma unroll
        for (int tn = 0; tn < 4; ++tn) {
            int r = wave_n * 64 + tn * 16 + lane16;
            bF[tn] = frag_from_lds(&sBt[buf][r][0], half);
        }

#pragma unroll
        for (int tm = 0; tm < 2; ++tm)
#pragma unroll
            for (int tn = 0; tn < 4; ++tn) {
                acc[tm][tn] = __builtin_amdgcn_wmma_f32_16x16x32_f16(
                    false, aHi[tm], false, bF[tn], (short)0, acc[tm][tn],
                    false, false);
                acc[tm][tn] = __builtin_amdgcn_wmma_f32_16x16x32_f16(
                    false, aLo[tm], false, bF[tn], (short)0, acc[tm][tn],
                    false, false);
            }

        if (it + 1 < NKSTEPS) ldsStore(buf ^ 1);
        __syncthreads();
    }

    float sp = softplusf(ScaleP[0]);
#pragma unroll
    for (int tm = 0; tm < 2; ++tm)
#pragma unroll
        for (int tn = 0; tn < 4; ++tn) {
            int col = n0 + wave_n * 64 + tn * 16 + lane16;
#pragma unroll
            for (int r = 0; r < 8; ++r) {
                int row = m0 + wave_m * 32 + tm * 16 + half * 8 + r;
                Out[(size_t)row * N_TOTAL + col] = acc[tm][tn][r] * sp;
            }
        }
}

extern "C" void kernel_launch(void* const* d_in, const int* in_sizes, int n_in,
                              void* d_out, int out_size, void* d_ws, size_t ws_size,
                              hipStream_t stream) {
    const float* x     = (const float*)d_in[0];   // (2,2048,4096) f32
    const float* sign  = (const float*)d_in[1];   // (4096,4096) f32, +/-1
    const float* scale = (const float*)d_in[2];   // scalar f32
    float*       out   = (float*)d_out;           // (2,2048,4096) f32

    const size_t nMK = (size_t)M_TOTAL * K_TOTAL;
    const size_t nKN = (size_t)K_TOTAL * N_TOTAL;
    const size_t need = (2 * nMK + nKN) * sizeof(_Float16);   // 96 MB

    dim3 ggrid(N_TOTAL / BN, M_TOTAL / BM);       // 32 x 32 workgroups

    if (ws_size >= need) {
        _Float16* xhi = (_Float16*)d_ws;
        _Float16* xlo = xhi + nMK;
        _Float16* sT  = xlo + nMK;
        ConvertX<<<(int)(nMK / 4 / 256), 256, 0, stream>>>(x, xhi, xlo);
        TransposeSign<<<dim3(N_TOTAL / 64, K_TOTAL / 64), 256, 0, stream>>>(sign, sT);
        GemmF16Wmma<<<ggrid, 256, 0, stream>>>(xhi, xlo, sT, scale, out);
    } else {
        GemmFusedWmma<<<ggrid, 256, 0, stream>>>(x, sign, scale, out);
    }
}
